// TimeDecayMultiHeadRichAttention_49039936586051
// MI455X (gfx1250) — matrix operand order repaired and checked by reading, not compiled, not run on hardware
//
#include <hip/hip_runtime.h>

typedef __bf16 bf16_t;
typedef __attribute__((ext_vector_type(16))) __bf16 v16bf;
typedef __attribute__((ext_vector_type(8)))  __bf16 v8bf;
typedef __attribute__((ext_vector_type(8)))  float  v8f;

#define B_   2048
#define S_   200
#define D_   128
#define H_   4
#define H1_  64
#define H2_  32
#define F_   512
#define ST_  13      // ceil(S/16)
#define SP_  208     // S padded to 16
#define TDEC 0.1f
#define NT_  512     // threads per block (16 waves)

// ---------------------------------------------------------------------------
// Pre-pack W1 (H,H1,F) f32 -> bf16 in exact WMMA B-fragment order.
// Fragment index: ((nt*16 + kk)*32 + lane), nt = h*4 + colchunk (0..15),
// kk = K-step (0..15). Each fragment-lane holds 16 contiguous bf16:
//   N = (nt&3)*16 + (lane&15),  K = kk*32 + (lane>>4)*8 + j + (j>=8 ? 8 : 0)   (B is 32x16/step)
// ---------------------------------------------------------------------------
__global__ void pack_w1(const float* __restrict__ W1, bf16_t* __restrict__ out) {
  int idx = blockIdx.x * blockDim.x + threadIdx.x;
  if (idx >= 16 * 16 * 32) return;
  int lane = idx & 31;
  int kk   = (idx >> 5) & 15;
  int nt   = idx >> 9;                 // 0..15 = h*4 + chunk
  int h    = nt >> 2;
  int n    = (nt & 3) * 16 + (lane & 15);
  int kr   = kk * 32 + (lane >> 4) * 8;
  const float* src = W1 + ((size_t)(h * H1_ + n)) * F_;
  bf16_t* dst = out + (size_t)idx * 16;
#pragma unroll
  for (int j = 0; j < 16; ++j) dst[j] = (bf16_t)src[kr + j + (j >= 8 ? 8 : 0)];
}

// W2 (H,H2,H1) -> fragments: ((h*2+nt)*2 + kk)*32 + lane
__global__ void pack_w2(const float* __restrict__ W2, bf16_t* __restrict__ out) {
  int idx = blockIdx.x * blockDim.x + threadIdx.x;
  if (idx >= 16 * 32) return;
  int lane = idx & 31;
  int kk   = (idx >> 5) & 1;
  int nt   = (idx >> 6) & 1;
  int h    = idx >> 7;
  int n    = nt * 16 + (lane & 15);
  int kr   = kk * 32 + (lane >> 4) * 8;
  const float* src = W2 + ((size_t)(h * H2_ + n)) * H1_;
  bf16_t* dst = out + (size_t)idx * 16;
#pragma unroll
  for (int j = 0; j < 16; ++j) dst[j] = (bf16_t)src[kr + j + (j >= 8 ? 8 : 0)];
}

// A-fragment loader: 16-bit A 16x32 layout — lane row = lane&15,
// halves 0..7  = K(kk*32 + (lane>>4)*8 + 0..7)
// halves 8..15 = K(kk*32 + 16 + (lane>>4)*8 + 0..7)  -> two ds_load_b128
__device__ __forceinline__ v16bf load_a_frag(const bf16_t* rowptr, int kbase) {
  v8bf lo = *(const v8bf*)(rowptr + kbase);
  v8bf hi = *(const v8bf*)(rowptr + kbase + 16);
  return __builtin_shufflevector(lo, hi, 0,1,2,3,4,5,6,7,8,9,10,11,12,13,14,15);
}
__device__ __forceinline__ v16bf load_b_frag(const bf16_t* p) {
  v8bf lo = *(const v8bf*)p;
  v8bf hi = *(const v8bf*)(p + 8);
  return __builtin_shufflevector(lo, hi, 0,1,2,3,4,5,6,7,8,9,10,11,12,13,14,15);
}

__global__ __launch_bounds__(NT_)
void attn_main(const float* __restrict__ query,
               const float* __restrict__ keys,
               const unsigned char* __restrict__ mask,
               const bf16_t* __restrict__ pW1,
               const float* __restrict__ b1,
               const float* __restrict__ a1,
               const bf16_t* __restrict__ pW2,
               const float* __restrict__ b2,
               const float* __restrict__ a2,
               const float* __restrict__ W3,
               const float* __restrict__ b3,
               const float* __restrict__ Wo,
               const float* __restrict__ bo,
               float* __restrict__ out,        // B*D
               float* __restrict__ out_avgw)   // B*S
{
  __shared__ bf16_t xs[16][F_];          // 16 KB  x-tile (bf16)
  __shared__ bf16_t h1s[16][H_ * H1_];   // 8 KB   layer-1 activations
  __shared__ float  h2s[16][H_ * H2_];   // 8 KB   layer-2 activations
  __shared__ float  sc[H_][SP_];         // scores / softmax weights
  __shared__ float  avgw[SP_];
  __shared__ float  qv[D_];
  __shared__ float  pcomb[4][D_];        // partial combined sums

  const int b    = blockIdx.x;
  const int tid  = threadIdx.x;
  const int lane = tid & 31;
  const int wave = tid >> 5;             // 0..15, == N-tile id (h*4 + chunk)
  const int row  = lane & 15;            // A row, D column
  const int koff = (lane >> 4) * 8;      // A K sub-offset

  // ---- register-resident B fragments: one N-tile per wave, whole K ----
  v16bf bw1[16];
#pragma unroll
  for (int kk = 0; kk < 16; ++kk)
    bw1[kk] = load_b_frag(pW1 + ((size_t)((wave * 16 + kk) * 32 + lane)) * 16);

  v16bf bw2[2];
  {
    int hm = (wave & 7) >> 1, ch = wave & 1;   // valid for waves 0..7
#pragma unroll
    for (int kk = 0; kk < 2; ++kk)
      bw2[kk] = load_b_frag(pW2 + ((size_t)(((hm * 2 + ch) * 2 + kk) * 32 + lane)) * 16);
  }

  if (tid < D_) qv[tid] = query[(size_t)b * D_ + tid];
  __syncthreads();

  for (int st = 0; st < ST_; ++st) {
    // ---- build x tile: [keys | q | keys*q | keys-q], bf16, rows s-tile ----
    for (int idx = tid; idx < 16 * F_; idx += NT_) {
      int r = idx >> 9;
      int f = idx & (F_ - 1);
      int s = st * 16 + r;
      float xvv = 0.f;
      if (s < S_) {
        int d   = f & (D_ - 1);
        int sel = f >> 7;
        float k = keys[((size_t)b * S_ + s) * D_ + d];
        float q = qv[d];
        xvv = (sel == 0) ? k : (sel == 1) ? q : (sel == 2) ? k * q : k - q;
      }
      xs[r][f] = (bf16_t)xvv;
    }
    // prefetch next s-tile's keys rows (overlaps with the GEMM below)
    if (st + 1 < ST_ && tid < 64) {
      int s = (st + 1) * 16 + (tid >> 2);
      if (s < S_)
        __builtin_prefetch(&keys[((size_t)b * S_ + s) * D_ + (tid & 3) * 32], 0, 1);
    }
    __syncthreads();

    // ---- GEMM1: h1 = leaky(x @ W1[h]^T + b1); M=16 N=256 K=512 ----
    {
      v8f acc = {};
#pragma unroll
      for (int kk = 0; kk < 16; ++kk) {
        v16bf a = load_a_frag(&xs[row][0], kk * 32 + koff);
        acc = __builtin_amdgcn_wmma_f32_16x16x32_bf16(
            false, a, false, bw1[kk], (short)0, acc, false, false);
      }
      int h    = wave >> 2;
      int col  = (wave & 3) * 16 + (lane & 15);
      float bv = b1[h * H1_ + col];
      float sl = a1[h];
#pragma unroll
      for (int r = 0; r < 8; ++r) {
        int m   = r + 8 * (lane >> 4);
        float v = acc[r] + bv;
        v = (v >= 0.f) ? v : sl * v;
        h1s[m][h * H1_ + col] = (bf16_t)v;
      }
    }
    __syncthreads();

    // ---- GEMM2: h2 = leaky(h1 @ W2[h]^T + b2); waves 0..7, one tile each --
    if (wave < 8) {
      int h = wave >> 1, ch = wave & 1;
      v8f acc = {};
#pragma unroll
      for (int kk = 0; kk < 2; ++kk) {
        v16bf a = load_a_frag(&h1s[row][0], h * H1_ + kk * 32 + koff);
        acc = __builtin_amdgcn_wmma_f32_16x16x32_bf16(
            false, a, false, bw2[kk], (short)0, acc, false, false);
      }
      int col  = ch * 16 + (lane & 15);
      float bv = b2[h * H2_ + col];
      float sl = a2[h];
#pragma unroll
      for (int r = 0; r < 8; ++r) {
        int m   = r + 8 * (lane >> 4);
        float v = acc[r] + bv;
        h2s[m][h * H2_ + col] = (v >= 0.f) ? v : sl * v;
      }
    }
    __syncthreads();

    // ---- scores = h2 . W3[h] + b3 ----
    if (tid < 64) {
      int r = tid & 15, h = tid >> 4;
      float accv = b3[h];
#pragma unroll
      for (int e = 0; e < H2_; ++e) accv += h2s[r][h * H2_ + e] * W3[h * H2_ + e];
      sc[h][st * 16 + r] = accv;
    }
    __syncthreads();
  }

  // ---- time decay + mask ----
  for (int idx = tid; idx < H_ * SP_; idx += NT_) {
    int h = idx / SP_, s = idx - h * SP_;
    float v = sc[h][s];
    if (s < S_ && mask[(size_t)b * S_ + s])
      v *= __expf(TDEC * (float)(s - S_ + 1));
    else
      v = -1.0e9f;
    sc[h][s] = v;
  }
  __syncthreads();

  // ---- softmax per head (one wave each, wave32 shfl reductions) ----
  if (wave < H_) {
    int h = wave;
    float mx = -3.4e38f;
    for (int s = lane; s < SP_; s += 32) mx = fmaxf(mx, sc[h][s]);
#pragma unroll
    for (int off = 16; off > 0; off >>= 1) mx = fmaxf(mx, __shfl_xor(mx, off));
    float sum = 0.f;
    for (int s = lane; s < SP_; s += 32) {
      float e = __expf(sc[h][s] - mx);
      sc[h][s] = e;
      sum += e;
    }
#pragma unroll
    for (int off = 16; off > 0; off >>= 1) sum += __shfl_xor(sum, off);
    float inv = 1.f / sum;
    for (int s = lane; s < SP_; s += 32) sc[h][s] *= inv;
  }
  __syncthreads();

  // ---- avg weights over heads (also = reduction weights for combined) ----
  for (int s = tid; s < SP_; s += NT_) {
    float v = 0.25f * (sc[0][s] + sc[1][s] + sc[2][s] + sc[3][s]);
    avgw[s] = v;
    if (s < S_) out_avgw[(size_t)b * S_ + s] = v;
  }
  __syncthreads();

  // ---- combined[d] = sum_s avgw[s] * keys[b,s,d]  (split over 4 groups) ----
  {
    int d = tid & (D_ - 1), grp = tid >> 7;       // 4 groups of 128 threads
    int s0 = grp * 50, s1 = s0 + 50;              // 4*50 = 200 = S
    float accv = 0.f;
    for (int s = s0; s < s1; ++s)
      accv += avgw[s] * keys[((size_t)b * S_ + s) * D_ + d];
    pcomb[grp][d] = accv;
  }
  __syncthreads();

  // ---- output = combined @ Wo^T + bo ----
  if (tid < D_) {
    // gather combined into registers via LDS partials
    float accv = bo[tid];
    const float* wr = Wo + (size_t)tid * D_;
#pragma unroll 4
    for (int j = 0; j < D_; ++j) {
      float cj = pcomb[0][j] + pcomb[1][j] + pcomb[2][j] + pcomb[3][j];
      accv += cj * wr[j];
    }
    out[(size_t)b * D_ + tid] = accv;
  }
}

extern "C" void kernel_launch(void* const* d_in, const int* in_sizes, int n_in,
                              void* d_out, int out_size, void* d_ws, size_t ws_size,
                              hipStream_t stream) {
  (void)in_sizes; (void)n_in; (void)out_size; (void)ws_size;
  const float* query = (const float*)d_in[0];
  const float* keys  = (const float*)d_in[1];
  const unsigned char* mask = (const unsigned char*)d_in[2];
  const float* W1 = (const float*)d_in[3];
  const float* b1 = (const float*)d_in[4];
  const float* a1 = (const float*)d_in[5];
  const float* W2 = (const float*)d_in[6];
  const float* b2 = (const float*)d_in[7];
  const float* a2 = (const float*)d_in[8];
  const float* W3 = (const float*)d_in[9];
  const float* b3 = (const float*)d_in[10];
  const float* Wo = (const float*)d_in[11];
  const float* bo = (const float*)d_in[12];

  float* out      = (float*)d_out;
  float* out_avgw = out + (size_t)B_ * D_;

  bf16_t* pW1 = (bf16_t*)d_ws;                       // 16*16*32*16 = 131072 elems (256 KB)
  bf16_t* pW2 = pW1 + (size_t)16 * 16 * 32 * 16;     // 16*32*16   = 8192 elems  (16 KB)

  pack_w1<<<(16 * 16 * 32 + 255) / 256, 256, 0, stream>>>(W1, pW1);
  pack_w2<<<2, 256, 0, stream>>>(W2, pW2);
  attn_main<<<B_, NT_, 0, stream>>>(query, keys, mask, pW1, b1, a1, pW2, b2, a2,
                                    W3, b3, Wo, bo, out, out_avgw);
}
